// FiLMRelationalMultiAggrMP_12403865551633
// MI455X (gfx1250) — matrix operand order repaired and checked by reading, not compiled, not run on hardware
//
#include <hip/hip_runtime.h>

// ---------------------------------------------------------------------------
// MI455X (gfx1250) FiLM/PNA relational multi-aggregation message passing.
//
// Roofline: edge GEMM = 157 GFLOP f32 -> bf16 hi/lo split on
// v_wmma_f32_16x16x32_bf16 (hi*hi + hi*lo + lo*hi ~ f32 accuracy).
// Dominant traffic was B-fragment L2 re-reads; this version register-blocks
// 4 edge tiles (64 edges) per workgroup so each B fragment loaded per kt is
// reused across 4 WMMA A tiles (B L2 traffic ~4.8 GB). A tiles are converted
// to bf16 hi/lo fragment layout ONCE at staging (stored directly in LDS),
// removing 8x-redundant per-wave cvt work from the inner loop.
// Aggregators (sum/mean-sum/var-sum/max) live directly in d_out[:, 0:512]
// via native f32 atomic-add / i32 atomic-max (valid: ReLU outputs >= 0).
// ---------------------------------------------------------------------------

typedef __attribute__((ext_vector_type(16))) __bf16 v16bf;
typedef __attribute__((ext_vector_type(8)))  float  v8f;

#define HDIM   128      // node feature width
#define KDIM   256      // 2*H
#define NCOLS  384      // 3*M
#define TILE_E 16       // edges per WMMA M tile
#define EPB    4        // edge tiles per block (64 edges)
#define NT_TOT 24       // 384/16 N-tiles
#define KT_TOT 8        // 256/32 K-steps
#define FRAG   512      // bf16 elems per 16x32 / 32x16 fragment

__device__ __forceinline__ unsigned short bf16_bits(__bf16 h) {
    return __builtin_bit_cast(unsigned short, h);
}

__device__ __forceinline__ void cvt_hilo(float f, unsigned short& h, unsigned short& l) {
    __bf16 hb = (__bf16)f;
    h = bf16_bits(hb);
    l = bf16_bits((__bf16)(f - (float)hb));
}

// ---------------------------------------------------------------------------
// Zero the aggregation region of d_out (cols 0..511 of each node) + deg.
// ---------------------------------------------------------------------------
__global__ void init_kernel(float* __restrict__ out, float* __restrict__ deg,
                            long long n512, int N) {
    long long idx = (long long)blockIdx.x * blockDim.x + threadIdx.x;
    if (idx < n512) {
        long long n = idx >> 9;
        int j = (int)(idx & 511);
        out[n * 1536 + j] = 0.0f;
    } else if (idx < n512 + N) {
        deg[idx - n512] = 0.0f;
    }
}

// ---------------------------------------------------------------------------
// Repack W (T,256,384) f32 into WMMA B-fragment-major bf16 hi/lo layout:
// fragment = 32x16 (KxN), lane l holds column n = nt*16+(l&15),
// element i <-> K = kt*32 + (l>=16 ? 16 : 0) + i.
// ---------------------------------------------------------------------------
__global__ void prep_w_kernel(const float* __restrict__ W,
                              unsigned short* __restrict__ whi,
                              unsigned short* __restrict__ wlo, int total) {
    int idx = blockIdx.x * blockDim.x + threadIdx.x;
    if (idx >= total) return;
    int pos = idx & (FRAG - 1);
    int fid = idx >> 9;
    int l  = pos >> 4;
    int i  = pos & 15;
    int kt = fid & (KT_TOT - 1);
    int ntt = fid >> 3;
    int nt = ntt % NT_TOT;
    int t  = ntt / NT_TOT;
    int K  = kt * 32 + ((l >> 4) << 4) + i;
    int n  = nt * 16 + (l & 15);
    float w = W[((long long)t * KDIM + K) * NCOLS + n];
    unsigned short h, lo;
    cvt_hilo(w, h, lo);
    whi[idx] = h;
    wlo[idx] = lo;
}

// ---------------------------------------------------------------------------
// Degree histogram over edge targets.
// ---------------------------------------------------------------------------
__global__ void deg_kernel(const long long* __restrict__ adj,
                           float* __restrict__ deg, int total) {
    int idx = blockIdx.x * blockDim.x + threadIdx.x;
    if (idx >= total) return;
    int tg = (int)adj[(long long)idx * 2 + 1];
    unsafeAtomicAdd(deg + tg, 1.0f);
}

// ---------------------------------------------------------------------------
// Cooperative A staging: gather 64 edge rows of [x[src] | x[tgt]] (256 f32),
// convert to bf16 hi/lo, and store directly in WMMA A-fragment layout in LDS.
// A-fragment element i of lane l (row m=l&15): K = kt*32 + 16*(l>>4) + i.
// A float4 at column c (c%4==0) maps to 4 consecutive
// fragment elements => one 8-byte LDS store each for hi and lo.
// ---------------------------------------------------------------------------
__device__ __forceinline__ void stage_a_tiles(
        const float* __restrict__ x,
        const int* srcs_s, const int* tgts_s,
        unsigned short* ahi_lds, unsigned short* alo_lds, int tid) {
    #pragma unroll
    for (int it = 0; it < 16; ++it) {
        int f   = tid + 256 * it;         // float4 id: 64 rows * 64 f4/row
        int row = f >> 6;
        int c   = (f & 63) << 2;          // column in floats
        const float* sp = (c < HDIM)
            ? (x + (long long)srcs_s[row] * HDIM + c)
            : (x + (long long)tgts_s[row] * HDIM + (c - HDIM));
        float4 v = *(const float4*)sp;

        int mt = row >> 4;
        int m  = row & 15;
        int kt = c >> 5;
        int kk = c & 31;
        int l  = ((kk >> 4) << 4) | m;
        int ib = kk & 15;                       // ibase, multiple of 4
        int off = (mt * KT_TOT + kt) * FRAG + l * 16 + ib;

        unsigned short h0, h1, h2, h3, l0, l1, l2, l3;
        cvt_hilo(v.x, h0, l0); cvt_hilo(v.y, h1, l1);
        cvt_hilo(v.z, h2, l2); cvt_hilo(v.w, h3, l3);
        uint2 hp, lp;
        hp.x = (unsigned)h0 | ((unsigned)h1 << 16);
        hp.y = (unsigned)h2 | ((unsigned)h3 << 16);
        lp.x = (unsigned)l0 | ((unsigned)l1 << 16);
        lp.y = (unsigned)l2 | ((unsigned)l3 << 16);
        *(uint2*)&ahi_lds[off] = hp;
        *(uint2*)&alo_lds[off] = lp;
    }
}

// ---------------------------------------------------------------------------
// Main GEMM: msgs = relu(feats @ W[t] + b[t]); scatter
//   cols [0,128)   -> atomic add  d_out[tgt, col]      (sum)
//   cols [128,256) -> atomic add  d_out[tgt, col]      (mean-sum)
//   cols [256,384) -> atomic max  d_out[tgt, col+128]  (max block @ 384..511)
// ---------------------------------------------------------------------------
__global__ __launch_bounds__(256)
void gemm_main_kernel(const float* __restrict__ x,
                      const float* __restrict__ bias,
                      const long long* __restrict__ adj,
                      const unsigned short* __restrict__ whi,
                      const unsigned short* __restrict__ wlo,
                      float* __restrict__ out,
                      int E, int blocks_per_t) {
    __shared__ unsigned short ahi_lds[EPB * KT_TOT * FRAG];  // 32 KB
    __shared__ unsigned short alo_lds[EPB * KT_TOT * FRAG];  // 32 KB
    __shared__ int srcs_s[EPB * TILE_E];
    __shared__ int tgts_s[EPB * TILE_E];

    const int tid = threadIdx.x;
    const int t  = blockIdx.x / blocks_per_t;
    const int e0 = (blockIdx.x % blocks_per_t) * (EPB * TILE_E);

    if (tid < EPB * TILE_E) {
        long long base = ((long long)t * E + e0 + tid) * 2;
        srcs_s[tid] = (int)adj[base + 0];
        tgts_s[tid] = (int)adj[base + 1];
    }
    __syncthreads();

    stage_a_tiles(x, srcs_s, tgts_s, ahi_lds, alo_lds, tid);
    __syncthreads();

    const int lane  = tid & 31;
    const int wave  = tid >> 5;
    const int m     = lane & 15;
    const int hlane = lane >> 4;

    v8f acc[EPB][3];
    #pragma unroll
    for (int mt = 0; mt < EPB; ++mt)
        #pragma unroll
        for (int j = 0; j < 3; ++j) acc[mt][j] = (v8f){};

    const unsigned short* whi_t = whi + (long long)t * NT_TOT * KT_TOT * FRAG;
    const unsigned short* wlo_t = wlo + (long long)t * NT_TOT * KT_TOT * FRAG;

    for (int kt = 0; kt < KT_TOT; ++kt) {
        // Hold this kt-step's 6 B fragments in registers (reused by 4 A tiles)
        v16bf bhi[3], blo[3];
        #pragma unroll
        for (int j = 0; j < 3; ++j) {
            int nt = wave * 3 + j;
            bhi[j] = *(const v16bf*)(whi_t + (nt * KT_TOT + kt) * FRAG + lane * 16);
            blo[j] = *(const v16bf*)(wlo_t + (nt * KT_TOT + kt) * FRAG + lane * 16);
        }
        #pragma unroll
        for (int mt = 0; mt < EPB; ++mt) {
            const int aoff = (mt * KT_TOT + kt) * FRAG + lane * 16;
            const v16bf ahi = *(const v16bf*)&ahi_lds[aoff];
            const v16bf alo = *(const v16bf*)&alo_lds[aoff];
            #pragma unroll
            for (int j = 0; j < 3; ++j) {
                v8f a = acc[mt][j];
                a = __builtin_amdgcn_wmma_f32_16x16x32_bf16(false, ahi, false, bhi[j], (short)0, a, false, false);
                a = __builtin_amdgcn_wmma_f32_16x16x32_bf16(false, ahi, false, blo[j], (short)0, a, false, false);
                a = __builtin_amdgcn_wmma_f32_16x16x32_bf16(false, alo, false, bhi[j], (short)0, a, false, false);
                acc[mt][j] = a;
            }
        }
    }

    // Epilogue: bias + relu + atomic scatter. C layout: lane holds column
    // n = nt*16 + (lane&15); VGPR v holds row v + 8*(lane>=16) of its tile.
    #pragma unroll
    for (int j = 0; j < 3; ++j) {
        int n = (wave * 3 + j) * 16 + m;          // global col 0..383
        float bv = bias[t * NCOLS + n];
        #pragma unroll
        for (int mt = 0; mt < EPB; ++mt) {
            #pragma unroll
            for (int v = 0; v < 8; ++v) {
                int row = mt * TILE_E + v + (hlane << 3);
                float val = fmaxf(acc[mt][j][v] + bv, 0.0f);
                float* outp = out + (long long)tgts_s[row] * 1536;
                if (n < 256) {
                    unsafeAtomicAdd(outp + n, val);           // sum / mean-sum
                } else {
                    atomicMax((int*)(outp + n + 128), __float_as_int(val));
                }
            }
        }
    }
}

// ---------------------------------------------------------------------------
// mean_agg = mean_sum / max(deg,1), in place at cols [128,256).
// ---------------------------------------------------------------------------
__global__ void mean_div_kernel(float* __restrict__ out,
                                const float* __restrict__ deg, long long total) {
    long long idx = (long long)blockIdx.x * blockDim.x + threadIdx.x;
    if (idx >= total) return;
    long long n = idx >> 7;
    int c = (int)(idx & 127);
    float d = fmaxf(deg[n], 1.0f);
    out[n * 1536 + 128 + c] *= (1.0f / d);
}

// ---------------------------------------------------------------------------
// Variance pass: recompute middle 128 columns (N-tile 8+wave per wave),
// var_e = relu(mm^2 - mean_agg[tgt]^2), atomic-add into cols [256,384).
// Per-edge +eps folded into finalize as deg*eps.
// ---------------------------------------------------------------------------
__global__ __launch_bounds__(256)
void gemm_var_kernel(const float* __restrict__ x,
                     const float* __restrict__ bias,
                     const long long* __restrict__ adj,
                     const unsigned short* __restrict__ whi,
                     const unsigned short* __restrict__ wlo,
                     float* __restrict__ out,
                     int E, int blocks_per_t) {
    __shared__ unsigned short ahi_lds[EPB * KT_TOT * FRAG];
    __shared__ unsigned short alo_lds[EPB * KT_TOT * FRAG];
    __shared__ int srcs_s[EPB * TILE_E];
    __shared__ int tgts_s[EPB * TILE_E];

    const int tid = threadIdx.x;
    const int t  = blockIdx.x / blocks_per_t;
    const int e0 = (blockIdx.x % blocks_per_t) * (EPB * TILE_E);

    if (tid < EPB * TILE_E) {
        long long base = ((long long)t * E + e0 + tid) * 2;
        srcs_s[tid] = (int)adj[base + 0];
        tgts_s[tid] = (int)adj[base + 1];
    }
    __syncthreads();

    stage_a_tiles(x, srcs_s, tgts_s, ahi_lds, alo_lds, tid);
    __syncthreads();

    const int lane  = tid & 31;
    const int wave  = tid >> 5;
    const int m     = lane & 15;
    const int hlane = lane >> 4;
    const int nt    = 8 + wave;                   // N-tiles 8..15 => cols 128..255

    v8f acc[EPB];
    #pragma unroll
    for (int mt = 0; mt < EPB; ++mt) acc[mt] = (v8f){};

    const unsigned short* whi_t = whi + (long long)t * NT_TOT * KT_TOT * FRAG;
    const unsigned short* wlo_t = wlo + (long long)t * NT_TOT * KT_TOT * FRAG;

    for (int kt = 0; kt < KT_TOT; ++kt) {
        const v16bf bhi = *(const v16bf*)(whi_t + (nt * KT_TOT + kt) * FRAG + lane * 16);
        const v16bf blo = *(const v16bf*)(wlo_t + (nt * KT_TOT + kt) * FRAG + lane * 16);
        #pragma unroll
        for (int mt = 0; mt < EPB; ++mt) {
            const int aoff = (mt * KT_TOT + kt) * FRAG + lane * 16;
            const v16bf ahi = *(const v16bf*)&ahi_lds[aoff];
            const v16bf alo = *(const v16bf*)&alo_lds[aoff];
            v8f a = acc[mt];
            a = __builtin_amdgcn_wmma_f32_16x16x32_bf16(false, ahi, false, bhi, (short)0, a, false, false);
            a = __builtin_amdgcn_wmma_f32_16x16x32_bf16(false, ahi, false, blo, (short)0, a, false, false);
            a = __builtin_amdgcn_wmma_f32_16x16x32_bf16(false, alo, false, bhi, (short)0, a, false, false);
            acc[mt] = a;
        }
    }

    int n = nt * 16 + m;                          // 128..255
    float bv = bias[t * NCOLS + n];
    #pragma unroll
    for (int mt = 0; mt < EPB; ++mt) {
        #pragma unroll
        for (int v = 0; v < 8; ++v) {
            int row = mt * TILE_E + v + (hlane << 3);
            float mm = fmaxf(acc[mt][v] + bv, 0.0f);
            float* outp = out + (long long)tgts_s[row] * 1536;
            float ma = outp[n];                   // finalized mean_agg
            float var = fmaxf(mm * mm - ma * ma, 0.0f);
            unsafeAtomicAdd(outp + n + 128, var); // var accum @ cols 256..383
        }
    }
}

// ---------------------------------------------------------------------------
// Finalize: std = sqrt(var_sum + deg*eps); write [out | amp*out | att*out].
// ---------------------------------------------------------------------------
__global__ void final_kernel(float* __restrict__ out,
                             const float* __restrict__ deg, long long total) {
    long long idx = (long long)blockIdx.x * blockDim.x + threadIdx.x;
    if (idx >= total) return;
    long long n = idx >> 9;
    int j = (int)(idx & 511);
    float dg = deg[n];
    long long base = n * 1536;
    float v = out[base + j];
    if (j >= 256 && j < 384) v = sqrtf(v + dg * 1e-7f);
    float ld  = logf(dg + 1.0f);
    float amp = ld * (1.0f / 1.1515f);
    float att = 1.1515f / (ld + 1e-7f);
    out[base + j]        = v;
    out[base + 512 + j]  = amp * v;
    out[base + 1024 + j] = att * v;
}

// ---------------------------------------------------------------------------
extern "C" void kernel_launch(void* const* d_in, const int* in_sizes, int n_in,
                              void* d_out, int out_size, void* d_ws, size_t ws_size,
                              hipStream_t stream) {
    const float*     x    = (const float*)d_in[0];
    const float*     W    = (const float*)d_in[1];
    const float*     bias = (const float*)d_in[2];
    const long long* adj  = (const long long*)d_in[3];
    float* out = (float*)d_out;

    const int N = in_sizes[0] / HDIM;
    const int T = in_sizes[2] / NCOLS;
    const int E = in_sizes[3] / (2 * T);
    const int blocks_per_t = E / (EPB * TILE_E);

    // Workspace: [deg: N f32][W_hi: T*24*8*512 bf16][W_lo: same]
    char* ws = (char*)d_ws;
    size_t degB = ((size_t)N * 4 + 4095) & ~(size_t)4095;
    size_t wElems = (size_t)T * NT_TOT * KT_TOT * FRAG;
    size_t wB = (wElems * 2 + 4095) & ~(size_t)4095;
    float*          deg = (float*)ws;
    unsigned short* whi = (unsigned short*)(ws + degB);
    unsigned short* wlo = (unsigned short*)(ws + degB + wB);

    const long long n512 = (long long)N * 512;
    {
        long long tot = n512 + N;
        init_kernel<<<(int)((tot + 255) / 256), 256, 0, stream>>>(out, deg, n512, N);
    }
    {
        int tot = (int)wElems;
        prep_w_kernel<<<(tot + 255) / 256, 256, 0, stream>>>(W, whi, wlo, tot);
    }
    {
        int tot = T * E;
        deg_kernel<<<(tot + 255) / 256, 256, 0, stream>>>(adj, deg, tot);
    }
    gemm_main_kernel<<<T * blocks_per_t, 256, 0, stream>>>(x, bias, adj, whi, wlo, out, E, blocks_per_t);
    {
        long long tot = (long long)N * 128;
        mean_div_kernel<<<(int)((tot + 255) / 256), 256, 0, stream>>>(out, deg, tot);
    }
    gemm_var_kernel<<<T * blocks_per_t, 256, 0, stream>>>(x, bias, adj, whi, wlo, out, E, blocks_per_t);
    final_kernel<<<(int)((n512 + 255) / 256), 256, 0, stream>>>(out, deg, n512);
}
